// RotaryYatAttention_19791209300373
// MI455X (gfx1250) — matrix operand order repaired and checked by reading, not compiled, not run on hardware
//
#include <hip/hip_runtime.h>
#include <hip/hip_bf16.h>

#define B_   2
#define S_   2048
#define D_   1024
#define H_   16
#define HD_  64
#define EPSF 1e-5f
#define CHUNK 128

typedef __attribute__((ext_vector_type(16))) __bf16 v16bf;
typedef __attribute__((ext_vector_type(8)))  __bf16 v8bf;
typedef __attribute__((ext_vector_type(8)))  float  v8f;

union Frag { v16bf v; v8bf h[2]; };

// ---- WMMA fragment index maps (CDNA5 ISA 7.12.2, wave32) ----
// A 16x32 (16-bit): lane holds row = lane&15; element e -> K = (e<8 ? hi*8+e : 16+hi*8+e-8)
// B 32x16 (16-bit): lane holds col = lane&15; this code loads element e -> K = hi*16+e (contiguous), as it does for A
// C/D 16x16 f32: VGPR j holds row j + (lane>>4)*8, col = lane&15.

// reductions across the 16-lane half of a wave32
__device__ __forceinline__ float redmax16(float v) {
  v = fmaxf(v, __shfl_xor(v, 1, 32));
  v = fmaxf(v, __shfl_xor(v, 2, 32));
  v = fmaxf(v, __shfl_xor(v, 4, 32));
  v = fmaxf(v, __shfl_xor(v, 8, 32));
  return v;
}
__device__ __forceinline__ float redsum16(float v) {
  v += __shfl_xor(v, 1, 32);
  v += __shfl_xor(v, 2, 32);
  v += __shfl_xor(v, 4, 32);
  v += __shfl_xor(v, 8, 32);
  return v;
}

// CDNA5 async global->LDS copy, 16 bytes per lane, tracked by ASYNCcnt.
__device__ __forceinline__ void async_ld16(uint32_t lds_off, const void* gptr) {
  asm volatile("global_load_async_to_lds_b128 %0, %1, off"
               :: "v"(lds_off), "v"((unsigned long long)(uintptr_t)gptr)
               : "memory");
}
__device__ __forceinline__ void wait_async0() {
  asm volatile("s_wait_asynccnt 0x0" ::: "memory");
}
__device__ __forceinline__ uint32_t lds_off_of(const void* p) {
  return (uint32_t)(uintptr_t)p;   // flat LDS aperture: addr[31:0] = wave-LDS offset
}

// ===================== one-time precision conversion =====================
// x (f32) -> Xb (bf16), elementwise
__global__ __launch_bounds__(256)
void cvt_bf16_kernel(const float* __restrict__ in, __bf16* __restrict__ out, int n) {
  const int i = (blockIdx.x * 256 + threadIdx.x) * 4;
  if (i + 3 < n) {
    const float4 f = *(const float4*)(in + i);
    out[i + 0] = (__bf16)f.x;
    out[i + 1] = (__bf16)f.y;
    out[i + 2] = (__bf16)f.z;
    out[i + 3] = (__bf16)f.w;
  }
}

// W[k][n] (f32) -> WT[n][k] (bf16), tiled transpose through LDS. grid (D/32, D/32).
__global__ __launch_bounds__(256)
void cvt_transpose_kernel(const float* __restrict__ W, __bf16* __restrict__ WT) {
  __shared__ float t[32][33];
  const int k0 = blockIdx.x * 32, n0 = blockIdx.y * 32;
  const int tid = threadIdx.x;
#pragma unroll
  for (int p = 0; p < 4; ++p) {
    const int i = tid + p * 256;
    const int row = i >> 5, col = i & 31;          // row=k, col=n (coalesced read)
    t[row][col] = W[(size_t)(k0 + row) * D_ + n0 + col];
  }
  __syncthreads();
#pragma unroll
  for (int p = 0; p < 4; ++p) {
    const int i = tid + p * 256;
    const int row = i >> 5, col = i & 31;          // row=n, col=k (coalesced write)
    WT[(size_t)(n0 + row) * D_ + k0 + col] = (__bf16)t[col][row];
  }
}

// ===================== fused projection (+RoPE, +norms) =====================
// grid: (B*S/128, H). 8 waves x 16 rows; B-tile (WT, [n][k] bf16) async double-buffered.
// doRope==1: Q/K -> Out[b][h][s][d] bf16 + Norms[b][h][s]
// doRope==0: V   -> Out[b][h][d][s] bf16 (TRANSPOSED for attention staging)
__global__ __launch_bounds__(256)
void yat_proj_kernel(const __bf16* __restrict__ Xb, const __bf16* __restrict__ WT,
                     __bf16* __restrict__ Out, float* __restrict__ Norms, int doRope)
{
  __shared__ __attribute__((aligned(16))) __bf16 lwT[2][64 * 32];  // [buf][n][k], 4KB each
  const int tid  = threadIdx.x;
  const int lane = tid & 31, wave = tid >> 5;
  const int r = lane & 15, hi = lane >> 4;
  const int rows0 = blockIdx.x * 128 + wave * 16;  // flat row into B*S
  const int head  = blockIdx.y;
  const int col0  = head * HD_;

  const uint32_t lw_l[2] = { lds_off_of(lwT[0]), lds_off_of(lwT[1]) };
  const int n_i = tid >> 2, c_i = tid & 3;                         // 64 rows x 4 chunks
  const uint32_t lds_dst = (uint32_t)(n_i * 64 + c_i * 16);
  const char* wrow = (const char*)(WT + (size_t)(col0 + n_i) * D_) + c_i * 16;

  v8f acc[4] = {};
  const __bf16* xr = Xb + (size_t)(rows0 + r) * D_;

  async_ld16(lw_l[0] + lds_dst, wrow);                             // prologue: tile k0=0
  int buf = 0;
  for (int k0 = 0; k0 < D_; k0 += 32) {
    wait_async0();
    __syncthreads();
    if (k0 + 32 < D_)
      async_ld16(lw_l[buf ^ 1] + lds_dst, wrow + (size_t)(k0 + 32) * 2);

    Frag a;
    a.h[0] = *(const v8bf*)(xr + k0 + hi * 16);
    a.h[1] = *(const v8bf*)(xr + k0 + hi * 16 + 8);
    const __bf16* lw = lwT[buf];
#pragma unroll
    for (int c = 0; c < 4; ++c) {
      const __bf16* wp = lw + (c * 16 + r) * 32 + hi * 16;
      Frag bfr;
      bfr.h[0] = *(const v8bf*)wp;
      bfr.h[1] = *(const v8bf*)(wp + 8);
      acc[c] = __builtin_amdgcn_wmma_f32_16x16x32_bf16(false, a.v, false, bfr.v,
                                                       (short)0, acc[c], false, false);
    }
    __syncthreads();
    buf ^= 1;
  }

  if (doRope) {
    float nsum[8] = {0.f, 0.f, 0.f, 0.f, 0.f, 0.f, 0.f, 0.f};
#pragma unroll
    for (int c = 0; c < 4; ++c) {
      const int d   = c * 16 + r;
      const int ii  = d >> 1;
      const bool odd = (d & 1) != 0;
      const float inv = __powf(10000.f, -(float)ii * (1.f / 32.f));
#pragma unroll
      for (int j = 0; j < 8; ++j) {
        const int rowg = rows0 + j + hi * 8;
        const int b = rowg >> 11, s = rowg & (S_ - 1);
        const float ang = (float)s * inv;
        float sn, cs;
        __sincosf(ang, &sn, &cs);
        const float v  = acc[c][j];
        const float pv = __shfl_xor(v, 1, 32);               // even<->odd partner
        const float res = odd ? fmaf(pv, sn, v * cs) : (v * cs - pv * sn);
        const __bf16 rb = (__bf16)res;
        Out[(((size_t)b * H_ + head) * S_ + s) * HD_ + d] = rb;
        const float rf = (float)rb;                          // norm of bf16-rounded value
        nsum[j] = fmaf(rf, rf, nsum[j]);
      }
    }
#pragma unroll
    for (int j = 0; j < 8; ++j) {
      const float t = redsum16(nsum[j]);
      const int rowg = rows0 + j + hi * 8;
      const int b = rowg >> 11, s = rowg & (S_ - 1);
      if (r == 0) Norms[((size_t)b * H_ + head) * S_ + s] = t;
    }
  } else {
    // V: store transposed [b][h][d][s]
#pragma unroll
    for (int c = 0; c < 4; ++c) {
      const int d = c * 16 + r;
#pragma unroll
      for (int j = 0; j < 8; ++j) {
        const int rowg = rows0 + j + hi * 8;
        const int b = rowg >> 11, s = rowg & (S_ - 1);
        Out[(((size_t)b * H_ + head) * HD_ + d) * S_ + s] = (__bf16)acc[c][j];
      }
    }
  }
}

// stage one K/V/kn chunk via async transfers (per-thread slices)
__device__ __forceinline__ void stage_kv(uint32_t kl, uint32_t vl, uint32_t nl,
                                         const char* kg, const char* vg, const char* ng,
                                         int tid) {
  for (int i = tid; i < (CHUNK * HD_ * 2) / 16; i += 256) {   // 1024 x 16B each
    async_ld16(kl + i * 16, kg + i * 16);                     // K contiguous
    const int row = i >> 4, cc = i & 15;                      // Vt: row=hd, 16B chunk cc
    async_ld16(vl + row * (CHUNK * 2) + cc * 16, vg + (size_t)row * (S_ * 2) + cc * 16);
  }
  if (tid < (CHUNK * 4) / 16) async_ld16(nl + tid * 16, ng + tid * 16);
}

// ===================== flash Yat-attention =====================
// grid: (B*H, S/128). 8 waves, each owns one 16-row q tile.
// K ([col][hd]) and Vt ([hd][col]) + kn async double-buffered in LDS.
__global__ __launch_bounds__(256)
void yat_attn_kernel(const __bf16* __restrict__ Qb, const __bf16* __restrict__ Kb,
                     const __bf16* __restrict__ Vt, const float* __restrict__ qnG,
                     const float* __restrict__ knG, const float* __restrict__ alphaG,
                     __bf16* __restrict__ Ob)
{
  __shared__ __attribute__((aligned(16))) __bf16 Kc[2][CHUNK * HD_];   // 2 x 16 KB
  __shared__ __attribute__((aligned(16))) __bf16 Vc[2][HD_ * CHUNK];   // 2 x 16 KB
  __shared__ __attribute__((aligned(16))) float  knc[2][CHUNK];        // 2 x 512 B
  __shared__ __attribute__((aligned(16))) __bf16 pstb[8][16 * 32];     // per-wave P, 8 KB

  const int tid  = threadIdx.x;
  const int lane = tid & 31, wave = tid >> 5;
  const int r = lane & 15, hi = lane >> 4;
  const int bh = blockIdx.x;
  const int qtile = blockIdx.y * 128 + wave * 16;
  const int b = bh >> 4, h = bh & 15;

  const float scale = __powf(8.0f / __logf(65.0f), alphaG[0]);  // (sqrt(64)/log(65))^alpha

  // Q fragments (K-dim 64 -> two 16x32 A fragments), loaded once, vectorized
  const __bf16* qrow = Qb + ((size_t)bh * S_ + qtile + r) * HD_;
  Frag a0, a1;
  a0.h[0] = *(const v8bf*)(qrow + hi * 16);
  a0.h[1] = *(const v8bf*)(qrow + hi * 16 + 8);
  a1.h[0] = *(const v8bf*)(qrow + 32 + hi * 16);
  a1.h[1] = *(const v8bf*)(qrow + 40 + hi * 16);

  float qnr[8];
#pragma unroll
  for (int j = 0; j < 8; ++j) qnr[j] = qnG[(size_t)bh * S_ + qtile + j + hi * 8];

  v8f acc[4] = {};
  float m[8], l[8];
#pragma unroll
  for (int j = 0; j < 8; ++j) { m[j] = -INFINITY; l[j] = 0.f; }

  const uint32_t kc_l[2] = { lds_off_of(Kc[0]),  lds_off_of(Kc[1])  };
  const uint32_t vc_l[2] = { lds_off_of(Vc[0]),  lds_off_of(Vc[1])  };
  const uint32_t kn_l[2] = { lds_off_of(knc[0]), lds_off_of(knc[1]) };
  const char* kgbase = (const char*)(Kb + (size_t)bh * S_ * HD_);
  const char* vgbase = (const char*)(Vt + (size_t)bh * HD_ * S_);
  const char* ngbase = (const char*)(knG + (size_t)bh * S_);

  const int nchunks = blockIdx.y + 1;     // causal: need k <= q0+127, CHUNK == 128
  stage_kv(kc_l[0], vc_l[0], kn_l[0], kgbase, vgbase, ngbase, tid);   // prologue
  int buf = 0;
  for (int ch = 0; ch < nchunks; ++ch) {
    const int kc0 = ch * CHUNK;
    wait_async0();
    __syncthreads();
    if (ch + 1 < nchunks) {
      const int kn0 = kc0 + CHUNK;
      stage_kv(kc_l[buf ^ 1], vc_l[buf ^ 1], kn_l[buf ^ 1],
               kgbase + (size_t)kn0 * HD_ * 2, vgbase + (size_t)kn0 * 2,
               ngbase + (size_t)kn0 * 4, tid);
    }

    const __bf16* Kcb = Kc[buf];
    const __bf16* Vcb = Vc[buf];
    const float*  ncb = knc[buf];

    for (int kcol0 = 0; kcol0 < CHUNK; kcol0 += 32) {
      if (kc0 + kcol0 <= qtile + 15) {   // wave-uniform causal skip
        // ---- scores: two 16-col tiles (QK^T, K-dim 64), vector LDS reads ----
        v8f c0 = {}, c1 = {};
        {
          const __bf16* kcp = Kcb + (kcol0 + r) * HD_ + hi * 16;
          Frag bk0, bk1;
          bk0.h[0] = *(const v8bf*)(kcp);      bk0.h[1] = *(const v8bf*)(kcp + 8);
          bk1.h[0] = *(const v8bf*)(kcp + 32); bk1.h[1] = *(const v8bf*)(kcp + 40);
          c0 = __builtin_amdgcn_wmma_f32_16x16x32_bf16(false, a0.v, false, bk0.v, (short)0, c0, false, false);
          c0 = __builtin_amdgcn_wmma_f32_16x16x32_bf16(false, a1.v, false, bk1.v, (short)0, c0, false, false);
        }
        {
          const __bf16* kcp = Kcb + (kcol0 + 16 + r) * HD_ + hi * 16;
          Frag bk0, bk1;
          bk0.h[0] = *(const v8bf*)(kcp);      bk0.h[1] = *(const v8bf*)(kcp + 8);
          bk1.h[0] = *(const v8bf*)(kcp + 32); bk1.h[1] = *(const v8bf*)(kcp + 40);
          c1 = __builtin_amdgcn_wmma_f32_16x16x32_bf16(false, a0.v, false, bk0.v, (short)0, c1, false, false);
          c1 = __builtin_amdgcn_wmma_f32_16x16x32_bf16(false, a1.v, false, bk1.v, (short)0, c1, false, false);
        }
        // ---- Yat score + causal mask + online softmax over 32 cols ----
        const float kn0 = ncb[kcol0 + r], kn1 = ncb[kcol0 + 16 + r];
        const int col0g = kc0 + kcol0 + r, col1g = col0g + 16;
        float corr[8];
#pragma unroll
        for (int j = 0; j < 8; ++j) {
          const int rowg = qtile + j + hi * 8;
          float qk = c0[j];
          float s0 = qk * qk / (qnr[j] + kn0 - 2.f * qk + EPSF) * scale;
          s0 = (col0g <= rowg) ? s0 : -INFINITY;
          qk = c1[j];
          float s1 = qk * qk / (qnr[j] + kn1 - 2.f * qk + EPSF) * scale;
          s1 = (col1g <= rowg) ? s1 : -INFINITY;
          const float mt = fmaxf(redmax16(fmaxf(s0, s1)), m[j]);
          const float cr = __expf(m[j] - mt);
          m[j] = mt;
          const __bf16 p0b = (__bf16)__expf(s0 - mt);
          const __bf16 p1b = (__bf16)__expf(s1 - mt);
          // normalizer from the same bf16-rounded P the WMMA consumes
          l[j] = l[j] * cr + redsum16((float)p0b + (float)p1b);
          corr[j] = cr;
          pstb[wave][(j + hi * 8) * 32 + r]      = p0b;   // transpose P through LDS
          pstb[wave][(j + hi * 8) * 32 + 16 + r] = p1b;
        }
        asm volatile("s_wait_dscnt 0x0" ::: "memory");    // P writes visible before reads
        // rescale running accumulator
#pragma unroll
        for (int cc = 0; cc < 4; ++cc)
#pragma unroll
          for (int j = 0; j < 8; ++j) acc[cc][j] *= corr[j];
        // ---- P (16x32) x V (32x64): 4 WMMAs, vector LDS reads ----
        Frag pa;
        const __bf16* pp = &pstb[wave][r * 32];
        pa.h[0] = *(const v8bf*)(pp + hi * 16);
        pa.h[1] = *(const v8bf*)(pp + hi * 16 + 8);
#pragma unroll
        for (int cc = 0; cc < 4; ++cc) {
          const __bf16* vp = Vcb + (cc * 16 + r) * CHUNK + kcol0 + hi * 16;
          Frag bv;
          bv.h[0] = *(const v8bf*)vp;
          bv.h[1] = *(const v8bf*)(vp + 8);
          acc[cc] = __builtin_amdgcn_wmma_f32_16x16x32_bf16(false, pa.v, false, bv.v,
                                                            (short)0, acc[cc], false, false);
        }
      }
    }
    __syncthreads();
    buf ^= 1;
  }

  // finalize: divide by row sum, store [b][s][h*64+d] as bf16 for the output GEMM
#pragma unroll
  for (int cc = 0; cc < 4; ++cc) {
    const int d = cc * 16 + r;
#pragma unroll
    for (int j = 0; j < 8; ++j) {
      const int srow = qtile + j + hi * 8;
      const float o = acc[cc][j] / l[j];
      Ob[((size_t)b * S_ + srow) * D_ + h * HD_ + d] = (__bf16)o;
    }
  }
}

// ===================== output projection =====================
// grid: (B*S/128, D/64). Same async double-buffered WT tiles, f32 output.
__global__ __launch_bounds__(256)
void yat_outproj_kernel(const __bf16* __restrict__ A, const __bf16* __restrict__ WT,
                        float* __restrict__ Out)
{
  __shared__ __attribute__((aligned(16))) __bf16 lwT[2][64 * 32];
  const int tid  = threadIdx.x;
  const int lane = tid & 31, wave = tid >> 5;
  const int r = lane & 15, hi = lane >> 4;
  const int rows0 = blockIdx.x * 128 + wave * 16;
  const int col0  = blockIdx.y * 64;

  const uint32_t lw_l[2] = { lds_off_of(lwT[0]), lds_off_of(lwT[1]) };
  const int n_i = tid >> 2, c_i = tid & 3;
  const uint32_t lds_dst = (uint32_t)(n_i * 64 + c_i * 16);
  const char* wrow = (const char*)(WT + (size_t)(col0 + n_i) * D_) + c_i * 16;

  v8f acc[4] = {};
  const __bf16* ar = A + (size_t)(rows0 + r) * D_;

  async_ld16(lw_l[0] + lds_dst, wrow);
  int buf = 0;
  for (int k0 = 0; k0 < D_; k0 += 32) {
    wait_async0();
    __syncthreads();
    if (k0 + 32 < D_)
      async_ld16(lw_l[buf ^ 1] + lds_dst, wrow + (size_t)(k0 + 32) * 2);

    Frag a;
    a.h[0] = *(const v8bf*)(ar + k0 + hi * 16);
    a.h[1] = *(const v8bf*)(ar + k0 + hi * 16 + 8);
    const __bf16* lw = lwT[buf];
#pragma unroll
    for (int c = 0; c < 4; ++c) {
      const __bf16* wp = lw + (c * 16 + r) * 32 + hi * 16;
      Frag bfr;
      bfr.h[0] = *(const v8bf*)wp;
      bfr.h[1] = *(const v8bf*)(wp + 8);
      acc[c] = __builtin_amdgcn_wmma_f32_16x16x32_bf16(false, a.v, false, bfr.v,
                                                       (short)0, acc[c], false, false);
    }
    __syncthreads();
    buf ^= 1;
  }
#pragma unroll
  for (int c = 0; c < 4; ++c)
#pragma unroll
    for (int j = 0; j < 8; ++j)
      Out[(size_t)(rows0 + j + hi * 8) * D_ + col0 + c * 16 + r] = acc[c][j];
}

// ===================== launch =====================
extern "C" void kernel_launch(void* const* d_in, const int* in_sizes, int n_in,
                              void* d_out, int out_size, void* d_ws, size_t ws_size,
                              hipStream_t stream) {
  (void)in_sizes; (void)n_in; (void)out_size; (void)ws_size;
  const float* x     = (const float*)d_in[0];
  const float* Wq    = (const float*)d_in[1];
  const float* Wk    = (const float*)d_in[2];
  const float* Wv    = (const float*)d_in[3];
  const float* Wo    = (const float*)d_in[4];
  const float* alpha = (const float*)d_in[5];

  char* ws = (char*)d_ws;
  const size_t bf16_qkv = (size_t)B_ * S_ * D_ * sizeof(__bf16);   // 8 MiB each
  const size_t bf16_w   = (size_t)D_ * D_ * sizeof(__bf16);        // 2 MiB each
  const size_t norm_sz  = (size_t)B_ * H_ * S_ * sizeof(float);    // 256 KiB each
  __bf16* Qb  = (__bf16*)ws;  ws += bf16_qkv;
  __bf16* Kb  = (__bf16*)ws;  ws += bf16_qkv;
  __bf16* Vt  = (__bf16*)ws;  ws += bf16_qkv;   // V transposed [b][h][hd][s]
  __bf16* Ob  = (__bf16*)ws;  ws += bf16_qkv;
  __bf16* Xb  = (__bf16*)ws;  ws += bf16_qkv;
  __bf16* WTq = (__bf16*)ws;  ws += bf16_w;
  __bf16* WTk = (__bf16*)ws;  ws += bf16_w;
  __bf16* WTv = (__bf16*)ws;  ws += bf16_w;
  __bf16* WTo = (__bf16*)ws;  ws += bf16_w;
  float*  qn  = (float*)ws;   ws += norm_sz;
  float*  kn  = (float*)ws;   ws += norm_sz;

  dim3 blk(256);
  // one-time converts
  const int nx = B_ * S_ * D_;
  cvt_bf16_kernel<<<dim3(nx / 1024), blk, 0, stream>>>(x, Xb, nx);
  dim3 gt(D_ / 32, D_ / 32);
  cvt_transpose_kernel<<<gt, blk, 0, stream>>>(Wq, WTq);
  cvt_transpose_kernel<<<gt, blk, 0, stream>>>(Wk, WTk);
  cvt_transpose_kernel<<<gt, blk, 0, stream>>>(Wv, WTv);
  cvt_transpose_kernel<<<gt, blk, 0, stream>>>(Wo, WTo);

  dim3 g1(B_ * S_ / 128, H_);
  yat_proj_kernel<<<g1, blk, 0, stream>>>(Xb, WTq, Qb, qn, 1);
  yat_proj_kernel<<<g1, blk, 0, stream>>>(Xb, WTk, Kb, kn, 1);
  yat_proj_kernel<<<g1, blk, 0, stream>>>(Xb, WTv, Vt, nullptr, 0);
  dim3 g2(B_ * H_, S_ / 128);
  yat_attn_kernel<<<g2, blk, 0, stream>>>(Qb, Kb, Vt, qn, kn, alpha, Ob);
  dim3 g3(B_ * S_ / 128, D_ / 64);
  yat_outproj_kernel<<<g3, blk, 0, stream>>>(Ob, WTo, (float*)d_out);
}
